// STTDecoder_5145370821528
// MI455X (gfx1250) — matrix operand order repaired and checked by reading, not compiled, not run on hardware
//
#include <hip/hip_runtime.h>

// ---------------------------------------------------------------------------
// MI455X (gfx1250) Tacotron-style attention decoder.
//
// - Weights + encoder memory converted once to bf16 (~54MB, L2-resident).
// - Prenet: two parallel WMMA GEMMs (M = T*B = 12800).
// - The T=400 sequential scan runs as ONE persistent kernel (68 WGs) with a
//   device-scope sense-reversal grid barrier: 3 barriers/step.
//     slot A: WGs 0..63 attention-LSTM(t);  WGs 64..67 outproj(t-1) (pipelined)
//     slot B: WGs 0..63 GMM attention(t)    (2 WGs per batch row, ENC split)
//     slot C: WGs 0..63 decoder-LSTM(t)
//   All recurrent GEMMs use v_wmma_f32_16x16x32_bf16 (wave32).
// ---------------------------------------------------------------------------

#define B_    32
#define L_    512
#define T_    400
#define EMB_  512
#define PRE_  256
#define ENC_  512
#define NH_   1024    // ARNN == DRNN
#define NSYM_ 256
#define NWG_  68      // persistent grid: 64 LSTM/attention + 4 outproj WGs

typedef unsigned short u16;
typedef __attribute__((ext_vector_type(16))) __bf16        v16bf;
typedef __attribute__((ext_vector_type(8)))  float         v8f;
typedef __attribute__((ext_vector_type(4)))  unsigned int  v4u;

union ABf { v4u q[2]; v16bf v; };

__device__ __forceinline__ u16 f2bf(float x) {
  unsigned int u = __float_as_uint(x);
  u += 0x7FFFu + ((u >> 16) & 1u);          // round-to-nearest-even
  return (u16)(u >> 16);
}
__device__ __forceinline__ float bf2f(u16 v) {
  return __uint_as_float(((unsigned)v) << 16);
}
__device__ __forceinline__ float sigm(float x) { return 1.0f / (1.0f + __expf(-x)); }

// Accumulate C += A[Mtile,K] * W[Ntile,K]^T over one K segment with bf16 WMMA.
__device__ __forceinline__ void wmma_seg(v8f& c,
                                         const u16* __restrict__ A, int lda, int mbase,
                                         const u16* __restrict__ W, int ldw, int rbase,
                                         int K, int lane) {
  const int m  = mbase + (lane & 15);
  const int r  = rbase + (lane & 15);
  const int hi = lane >> 4;
  const u16* arow = A + (size_t)m * lda;
  const u16* wrow = W + (size_t)r * ldw;
  for (int k = 0; k < K; k += 32) {
    ABf a, b;
    // A 16x32 bf16: lanes0-15 K={0..7,16..23}, lanes16-31 K={8..15,24..31}
    a.q[0] = *(const v4u*)(arow + k + hi * 8);
    a.q[1] = *(const v4u*)(arow + k + 16 + hi * 8);
    // B 32x16 bf16: lanes0-15 K={0..7,16..23}, lanes16-31 K={8..15,24..31} (N = lane&15)
    b.q[0] = *(const v4u*)(wrow + k + hi * 8);
    b.q[1] = *(const v4u*)(wrow + k + 16 + hi * 8);
    c = __builtin_amdgcn_wmma_f32_16x16x32_bf16(false, a.v, false, b.v,
                                                (short)0, c, false, false);
  }
}

// Device-scope sense-reversal grid barrier (68 WGs are trivially co-resident).
__device__ __forceinline__ void grid_barrier(unsigned* cnt, unsigned* gen) {
  __threadfence();                 // release my global stores to device scope
  __syncthreads();
  if (threadIdx.x == 0) {
    unsigned g = __hip_atomic_load(gen, __ATOMIC_RELAXED, __HIP_MEMORY_SCOPE_AGENT);
    unsigned arrived =
        __hip_atomic_fetch_add(cnt, 1u, __ATOMIC_ACQ_REL, __HIP_MEMORY_SCOPE_AGENT);
    if (arrived == NWG_ - 1) {
      __hip_atomic_store(cnt, 0u, __ATOMIC_RELAXED, __HIP_MEMORY_SCOPE_AGENT);
      __hip_atomic_store(gen, g + 1u, __ATOMIC_RELEASE, __HIP_MEMORY_SCOPE_AGENT);
    } else {
      while (__hip_atomic_load(gen, __ATOMIC_ACQUIRE, __HIP_MEMORY_SCOPE_AGENT) == g)
        __builtin_amdgcn_s_sleep(1);
    }
  }
  __syncthreads();
  __threadfence();                 // acquire: invalidate stale near caches
}

// ---------------------------------------------------------------------------
// One-time preprocessing kernels
// ---------------------------------------------------------------------------
__global__ void k_f32_to_bf16(const float* __restrict__ s, u16* __restrict__ d, int n) {
  int i = blockIdx.x * 256 + threadIdx.x;
  if (i < n) d[i] = f2bf(s[i]);
}

// decoder_inputs (B, EMB, T) -> shifted x (T, B, EMB) bf16; x[0]=0, x[t]=din[:,:,t-1]
__global__ void k_prep_x(const float* __restrict__ din, u16* __restrict__ xbf) {
  int id = blockIdx.x * 256 + threadIdx.x;            // id = b*EMB*T + e*T + t
  if (id >= B_ * EMB_ * T_) return;
  int t = id % T_;
  int e = (id / T_) % EMB_;
  int b = id / (T_ * EMB_);
  float v = (t == 0) ? 0.0f : din[id - 1];
  xbf[(size_t)t * (B_ * EMB_) + b * EMB_ + e] = f2bf(v);
}

// C = relu(A[M,K] @ W[N,K]^T + bias) -> bf16 (prenet layers).
// grid: (M/32, N/64), 256 threads (8 waves: 2 M-tiles x 4 N-tiles).
__global__ void k_gemm_bias_relu(const u16* __restrict__ A, int lda,
                                 const u16* __restrict__ W, int ldw,
                                 const float* __restrict__ bias,
                                 u16* __restrict__ C, int ldc, int K) {
  const int tid  = threadIdx.x;
  const int lane = tid & 31;
  const int wave = tid >> 5;
  const int mt = wave & 1, nt = wave >> 1;
  const int mbase = blockIdx.x * 32 + mt * 16;
  const int nbase = blockIdx.y * 64 + nt * 16;
  v8f c = {0.f, 0.f, 0.f, 0.f, 0.f, 0.f, 0.f, 0.f};
  wmma_seg(c, A, lda, mbase, W, ldw, nbase, K, lane);
  const int hi = lane >> 4;
  const int n  = nbase + (lane & 15);
  const float bv = bias[n];
#pragma unroll
  for (int j = 0; j < 8; ++j) {
    int m = mbase + j + hi * 8;
    float v = c[j] + bv;
    v = fmaxf(v, 0.0f);
    C[(size_t)m * ldc + n] = f2bf(v);
  }
}

// ---------------------------------------------------------------------------
// Persistent decoder scan kernel: grid = NWG_ x 256 threads.
// ---------------------------------------------------------------------------
__global__ void __launch_bounds__(256)
k_decoder_scan(const u16* __restrict__ P2,
               const u16* __restrict__ aWih_bf, const u16* __restrict__ aWhh_bf,
               const u16* __restrict__ dWih_bf, const u16* __restrict__ dWhh_bf,
               const u16* __restrict__ oW_bf,
               const float* __restrict__ abih, const float* __restrict__ abhh,
               const float* __restrict__ dbih, const float* __restrict__ dbhh,
               const float* __restrict__ attWp, const float* __restrict__ attbp,
               const float* __restrict__ ob,
               const u16* __restrict__ mem_bf, const int* __restrict__ memlen,
               u16* __restrict__ ah0, u16* __restrict__ ah1,
               u16* __restrict__ dh0, u16* __restrict__ dh1,
               float* __restrict__ ah_f32, float* __restrict__ ac,
               float* __restrict__ dc, u16* __restrict__ ctx_bf,
               float* __restrict__ mean0, float* __restrict__ mean1,
               float* __restrict__ logits, float* __restrict__ aligns,
               float* __restrict__ ps,
               unsigned* bar_cnt, unsigned* bar_gen) {
  __shared__ float g4[4][B_][16];          // LSTM gate exchange (8KB)
  __shared__ float sred[8][32][8];         // attention ctx partials (8KB)
  __shared__ float red0[256], red1[256];   // attention param reductions
  __shared__ float sw[L_];                 // attention weights
  __shared__ float smean, sscale;

  const int wg   = blockIdx.x;
  const int tid  = threadIdx.x;
  const int lane = tid & 31;
  const int wave = tid >> 5;
  const int gi = wave >> 1;                // gate index for LSTM phases
  const int mt = wave & 1;                 // batch half
  const int hi = lane >> 4;
  const int nl = lane & 15;

  u16*   ahb[2] = {ah0, ah1};
  u16*   dhb[2] = {dh0, dh1};
  float* mnb[2] = {mean0, mean1};

  for (int t = 0; t < T_; ++t) {
    u16* ah_rd = ahb[t & 1];
    u16* ah_wr = ahb[(t + 1) & 1];
    u16* dh_rd = dhb[t & 1];
    u16* dh_wr = dhb[(t + 1) & 1];
    float* mean_rd = mnb[t & 1];
    float* mean_wr = mnb[(t + 1) & 1];
    const u16* xt = P2 + (size_t)t * B_ * PRE_;

    // ------- slot A: attention LSTM (WGs 0..63) || outproj(t-1) (WGs 64..67)
    if (wg < 64) {
      const int nb = wg * 16;
      const int rbase = gi * NH_ + nb;
      v8f c = {0.f, 0.f, 0.f, 0.f, 0.f, 0.f, 0.f, 0.f};
      wmma_seg(c, xt,     PRE_, mt * 16, aWih_bf,        PRE_ + ENC_, rbase, PRE_, lane);
      wmma_seg(c, ctx_bf, ENC_, mt * 16, aWih_bf + PRE_, PRE_ + ENC_, rbase, ENC_, lane);
      wmma_seg(c, ah_rd,  NH_,  mt * 16, aWhh_bf,        NH_,         rbase, NH_,  lane);
#pragma unroll
      for (int j = 0; j < 8; ++j) g4[gi][mt * 16 + hi * 8 + j][nl] = c[j];
      __syncthreads();
#pragma unroll
      for (int s = 0; s < 2; ++s) {
        int item = tid + s * 256;
        int b = item >> 4, nn = item & 15;
        int col = nb + nn;
        float iv = g4[0][b][nn] + abih[0 * NH_ + col] + abhh[0 * NH_ + col];
        float fv = g4[1][b][nn] + abih[1 * NH_ + col] + abhh[1 * NH_ + col];
        float gv = g4[2][b][nn] + abih[2 * NH_ + col] + abhh[2 * NH_ + col];
        float ov = g4[3][b][nn] + abih[3 * NH_ + col] + abhh[3 * NH_ + col];
        float cold = ac[(size_t)b * NH_ + col];
        float cn = sigm(fv) * cold + sigm(iv) * tanhf(gv);
        float h  = sigm(ov) * tanhf(cn);
        ac[(size_t)b * NH_ + col]     = cn;
        ah_wr[(size_t)b * NH_ + col]  = f2bf(h);
        ah_f32[(size_t)b * NH_ + col] = h;
      }
      __syncthreads();
    } else if (t > 0) {
      // pipelined output projection for step t-1: uses dh(t-1)=dh_rd, ctx(t-1)
      const int nt = wave >> 1;
      const int nbase = (wg - 64) * 64 + nt * 16;
      v8f c = {0.f, 0.f, 0.f, 0.f, 0.f, 0.f, 0.f, 0.f};
      wmma_seg(c, dh_rd,  NH_,  mt * 16, oW_bf,       NH_ + ENC_, nbase, NH_,  lane);
      wmma_seg(c, ctx_bf, ENC_, mt * 16, oW_bf + NH_, NH_ + ENC_, nbase, ENC_, lane);
      const int n = nbase + nl;
      const float bv = ob[n];
#pragma unroll
      for (int j = 0; j < 8; ++j) {
        int b = mt * 16 + j + hi * 8;
        logits[(size_t)b * (T_ * NSYM_) + (size_t)(t - 1) * NSYM_ + n] = c[j] + bv;
      }
    }
    grid_barrier(bar_cnt, bar_gen);

    // ------- slot B: GMM attention (WGs 0..63; 2 WGs per batch, ENC split) --
    if (wg < 2 * B_) {
      const int b = wg >> 1;
      const int half = wg & 1;
      const float* ah = ah_f32 + (size_t)b * NH_;
      float p0 = 0.f, p1 = 0.f;
      for (int k = tid; k < NH_; k += 256) {
        float h = ah[k];
        p0 += h * attWp[k];
        p1 += h * attWp[NH_ + k];
      }
      red0[tid] = p0; red1[tid] = p1;
      __syncthreads();
      for (int s = 128; s > 0; s >>= 1) {
        if (tid < s) { red0[tid] += red0[tid + s]; red1[tid] += red1[tid + s]; }
        __syncthreads();
      }
      if (tid == 0) {
        float pp0 = red0[0] + attbp[0];
        float pp1 = red1[0] + attbp[1];
        float m = mean_rd[b] + __expf(pp0) * 1.0f;      // MEAN_COEFF
        smean = m;
        sscale = __expf(pp1) * 10.0f;                   // SCALE_COEFF
        if (half == 0) {
          mean_wr[b] = m;
          ps[(size_t)b * (T_ * 2) + t * 2 + 0] = pp0;
          ps[(size_t)b * (T_ * 2) + t * 2 + 1] = pp1;
        }
      }
      __syncthreads();
      const float mean = smean, scale = sscale;
      const float Z = sqrtf(2.0f * 3.14159265358979323f * scale * scale);
      const int ml = memlen[b];
      for (int l = tid; l < L_; l += 256) {
        float d = (float)l - mean;
        float w = __expf(-0.5f * d * d / (scale * scale)) / Z;
        if (l >= ml) w = 0.0f;
        sw[l] = w;
        if (half == 0)
          aligns[(size_t)b * (T_ * L_) + (size_t)t * L_ + l] = w;
      }
      __syncthreads();
      // ctx partial: this WG covers d in [half*256, half*256+256)
      const int dg = tid & 31;               // 32 groups of 8 contiguous d
      const int lc = tid >> 5;               // 8 chunks of 64 l rows
      const int d0 = half * 256 + dg * 8;
      float acc[8] = {0.f, 0.f, 0.f, 0.f, 0.f, 0.f, 0.f, 0.f};
      const u16* mb = mem_bf + (size_t)b * L_ * ENC_ + d0;
      for (int l = lc * 64; l < lc * 64 + 64; ++l) {
        float w = sw[l];
        v4u q = *(const v4u*)(mb + (size_t)l * ENC_);   // 8 bf16
#pragma unroll
        for (int j = 0; j < 4; ++j) {
          unsigned u = q[j];
          acc[2 * j]     += w * bf2f((u16)(u & 0xffffu));
          acc[2 * j + 1] += w * bf2f((u16)(u >> 16));
        }
      }
#pragma unroll
      for (int j = 0; j < 8; ++j) sred[lc][dg][j] = acc[j];
      __syncthreads();
      {                                       // reduce 8 l-chunks -> 256 cols
        int dg2 = tid >> 3;                   // 0..31
        int dj  = tid & 7;
        float s = 0.f;
#pragma unroll
        for (int c2 = 0; c2 < 8; ++c2) s += sred[c2][dg2][dj];
        ctx_bf[(size_t)b * ENC_ + half * 256 + dg2 * 8 + dj] = f2bf(s);
      }
    }
    grid_barrier(bar_cnt, bar_gen);

    // ------- slot C: decoder LSTM (WGs 0..63) ------------------------------
    if (wg < 64) {
      const int nb = wg * 16;
      const int rbase = gi * NH_ + nb;
      v8f c = {0.f, 0.f, 0.f, 0.f, 0.f, 0.f, 0.f, 0.f};
      wmma_seg(c, ah_wr,  NH_,  mt * 16, dWih_bf,       NH_ + ENC_, rbase, NH_,  lane);
      wmma_seg(c, ctx_bf, ENC_, mt * 16, dWih_bf + NH_, NH_ + ENC_, rbase, ENC_, lane);
      wmma_seg(c, dh_rd,  NH_,  mt * 16, dWhh_bf,       NH_,        rbase, NH_,  lane);
#pragma unroll
      for (int j = 0; j < 8; ++j) g4[gi][mt * 16 + hi * 8 + j][nl] = c[j];
      __syncthreads();
#pragma unroll
      for (int s = 0; s < 2; ++s) {
        int item = tid + s * 256;
        int b = item >> 4, nn = item & 15;
        int col = nb + nn;
        float iv = g4[0][b][nn] + dbih[0 * NH_ + col] + dbhh[0 * NH_ + col];
        float fv = g4[1][b][nn] + dbih[1 * NH_ + col] + dbhh[1 * NH_ + col];
        float gv = g4[2][b][nn] + dbih[2 * NH_ + col] + dbhh[2 * NH_ + col];
        float ov = g4[3][b][nn] + dbih[3 * NH_ + col] + dbhh[3 * NH_ + col];
        float cold = dc[(size_t)b * NH_ + col];
        float cn = sigm(fv) * cold + sigm(iv) * tanhf(gv);
        float h  = sigm(ov) * tanhf(cn);
        dc[(size_t)b * NH_ + col]    = cn;
        dh_wr[(size_t)b * NH_ + col] = f2bf(h);
      }
      __syncthreads();
    }
    grid_barrier(bar_cnt, bar_gen);
  }

  // drain the software pipeline: outproj for the final step (T-1)
  if (wg >= 64) {
    u16* dh_last = dhb[T_ & 1];             // dh_wr of step T-1
    const int nt = wave >> 1;
    const int nbase = (wg - 64) * 64 + nt * 16;
    v8f c = {0.f, 0.f, 0.f, 0.f, 0.f, 0.f, 0.f, 0.f};
    wmma_seg(c, dh_last, NH_,  mt * 16, oW_bf,       NH_ + ENC_, nbase, NH_,  lane);
    wmma_seg(c, ctx_bf,  ENC_, mt * 16, oW_bf + NH_, NH_ + ENC_, nbase, ENC_, lane);
    const int n = nbase + nl;
    const float bv = ob[n];
#pragma unroll
    for (int j = 0; j < 8; ++j) {
      int b = mt * 16 + j + hi * 8;
      logits[(size_t)b * (T_ * NSYM_) + (size_t)(T_ - 1) * NSYM_ + n] = c[j] + bv;
    }
  }
}

// ---------------------------------------------------------------------------
// Host launch
// ---------------------------------------------------------------------------
extern "C" void kernel_launch(void* const* d_in, const int* in_sizes, int n_in,
                              void* d_out, int out_size, void* d_ws, size_t ws_size,
                              hipStream_t stream) {
  (void)in_sizes; (void)n_in; (void)out_size; (void)ws_size;

  const float* memory  = (const float*)d_in[0];
  const float* din     = (const float*)d_in[1];
  const int*   memlen  = (const int*)d_in[2];
  const float* pW1     = (const float*)d_in[3];
  const float* pb1     = (const float*)d_in[4];
  const float* pW2     = (const float*)d_in[5];
  const float* pb2     = (const float*)d_in[6];
  const float* aWih    = (const float*)d_in[7];
  const float* aWhh    = (const float*)d_in[8];
  const float* abih    = (const float*)d_in[9];
  const float* abhh    = (const float*)d_in[10];
  const float* attWp   = (const float*)d_in[11];
  const float* attbp   = (const float*)d_in[12];
  const float* dWih    = (const float*)d_in[13];
  const float* dWhh    = (const float*)d_in[14];
  const float* dbih    = (const float*)d_in[15];
  const float* dbhh    = (const float*)d_in[16];
  const float* oW      = (const float*)d_in[17];
  const float* ob      = (const float*)d_in[18];

  float* logits = (float*)d_out;                         // (B,T,NSYM)
  float* aligns = logits + (size_t)B_ * T_ * NSYM_;      // (B,T,L)
  float* ps     = aligns + (size_t)B_ * T_ * L_;         // (B,T,2)

  size_t off = 0;
  auto carve = [&](size_t bytes) -> char* {
    char* p = (char*)d_ws + off;
    off = (off + bytes + 255) & ~(size_t)255;
    return p;
  };
  u16* aWih_bf = (u16*)carve((size_t)4 * NH_ * (PRE_ + ENC_) * 2);
  u16* aWhh_bf = (u16*)carve((size_t)4 * NH_ * NH_ * 2);
  u16* dWih_bf = (u16*)carve((size_t)4 * NH_ * (NH_ + ENC_) * 2);
  u16* dWhh_bf = (u16*)carve((size_t)4 * NH_ * NH_ * 2);
  u16* oW_bf   = (u16*)carve((size_t)NSYM_ * (NH_ + ENC_) * 2);
  u16* pW1_bf  = (u16*)carve((size_t)PRE_ * EMB_ * 2);
  u16* pW2_bf  = (u16*)carve((size_t)PRE_ * PRE_ * 2);
  u16* mem_bf  = (u16*)carve((size_t)B_ * L_ * ENC_ * 2);
  u16* xbf     = (u16*)carve((size_t)T_ * B_ * EMB_ * 2);
  u16* P1      = (u16*)carve((size_t)T_ * B_ * PRE_ * 2);
  u16* P2      = (u16*)carve((size_t)T_ * B_ * PRE_ * 2);
  u16* ah_bf0  = (u16*)carve((size_t)B_ * NH_ * 2);
  u16* ah_bf1  = (u16*)carve((size_t)B_ * NH_ * 2);
  u16* dh_bf0  = (u16*)carve((size_t)B_ * NH_ * 2);
  u16* dh_bf1  = (u16*)carve((size_t)B_ * NH_ * 2);
  float* ah_f32 = (float*)carve((size_t)B_ * NH_ * 4);
  float* ac     = (float*)carve((size_t)B_ * NH_ * 4);
  float* dc     = (float*)carve((size_t)B_ * NH_ * 4);
  u16* ctx_bf   = (u16*)carve((size_t)B_ * ENC_ * 2);
  float* mean0  = (float*)carve((size_t)B_ * 4);
  float* mean1  = (float*)carve((size_t)B_ * 4);
  unsigned* bar_cnt = (unsigned*)carve(256);
  unsigned* bar_gen = (unsigned*)carve(256);

  auto conv = [&](const float* s, u16* d, size_t n) {
    k_f32_to_bf16<<<(unsigned)((n + 255) / 256), 256, 0, stream>>>(s, d, (int)n);
  };
  conv(aWih, aWih_bf, (size_t)4 * NH_ * (PRE_ + ENC_));
  conv(aWhh, aWhh_bf, (size_t)4 * NH_ * NH_);
  conv(dWih, dWih_bf, (size_t)4 * NH_ * (NH_ + ENC_));
  conv(dWhh, dWhh_bf, (size_t)4 * NH_ * NH_);
  conv(oW,   oW_bf,   (size_t)NSYM_ * (NH_ + ENC_));
  conv(pW1,  pW1_bf,  (size_t)PRE_ * EMB_);
  conv(pW2,  pW2_bf,  (size_t)PRE_ * PRE_);
  conv(memory, mem_bf, (size_t)B_ * L_ * ENC_);

  {
    size_t n = (size_t)B_ * EMB_ * T_;
    k_prep_x<<<(unsigned)((n + 255) / 256), 256, 0, stream>>>(din, xbf);
  }
  {
    dim3 g1(T_ * B_ / 32, PRE_ / 64);
    k_gemm_bias_relu<<<g1, 256, 0, stream>>>(xbf, EMB_, pW1_bf, EMB_, pb1, P1, PRE_, EMB_);
    k_gemm_bias_relu<<<g1, 256, 0, stream>>>(P1, PRE_, pW2_bf, PRE_, pb2, P2, PRE_, PRE_);
  }

  // zero recurrent state + barrier state (every call -> deterministic replay)
  hipMemsetAsync(ah_bf0, 0, (size_t)B_ * NH_ * 2, stream);
  hipMemsetAsync(dh_bf0, 0, (size_t)B_ * NH_ * 2, stream);
  hipMemsetAsync(ac, 0, (size_t)B_ * NH_ * 4, stream);
  hipMemsetAsync(dc, 0, (size_t)B_ * NH_ * 4, stream);
  hipMemsetAsync(ctx_bf, 0, (size_t)B_ * ENC_ * 2, stream);
  hipMemsetAsync(mean0, 0, (size_t)B_ * 4, stream);
  hipMemsetAsync(bar_cnt, 0, 256, stream);
  hipMemsetAsync(bar_gen, 0, 256, stream);

  // one persistent kernel runs the whole 400-step scan
  k_decoder_scan<<<NWG_, 256, 0, stream>>>(
      P2, aWih_bf, aWhh_bf, dWih_bf, dWhh_bf, oW_bf,
      abih, abhh, dbih, dbhh, attWp, attbp, ob,
      mem_bf, memlen,
      ah_bf0, ah_bf1, dh_bf0, dh_bf1,
      ah_f32, ac, dc, ctx_bf, mean0, mean1,
      logits, aligns, ps,
      bar_cnt, bar_gen);
}
